// ProxyAttention_26809185862108
// MI455X (gfx1250) — matrix operand order repaired and checked by reading, not compiled, not run on hardware
//
#include <hip/hip_runtime.h>

// ---------------- problem constants ----------------
#define DIMC   1024
#define HEADS  16
#define HD     64
#define NREAL  819      // real cluster tokens
#define NPAD   832      // 819 padded to multiple of 16 (and of 64)
#define LLEN   1024     // proxy length
#define BATCH  8
#define MREAL  6552     // B*N
#define MQ     6560     // padded to multiple of 32 (205 tiles of 32)
#define ATT_SCALE 0.125f

typedef __attribute__((ext_vector_type(16))) __bf16 bf16x16;
typedef __attribute__((ext_vector_type(8)))  float  f32x8;

union FragCvt {
  bf16x16        bf;
  unsigned short us[16];
  uint4          q[2];
};

__device__ __forceinline__ f32x8 zero8() {
  f32x8 z = {0.f,0.f,0.f,0.f,0.f,0.f,0.f,0.f};
  return z;
}

// fp32 -> bf16, round-to-nearest-even
__device__ __forceinline__ unsigned short f2bf(float f) {
  unsigned int u = __float_as_uint(f);
  u += 0x7FFFu + ((u >> 16) & 1u);
  return (unsigned short)(u >> 16);
}

__device__ __forceinline__ f32x8 wmma_bf16(bf16x16 a, bf16x16 b, f32x8 c) {
  // D = A(16x32) * B(32x16) + C, f32 accumulate
  return __builtin_amdgcn_wmma_f32_16x16x32_bf16(false, a, false, b, (short)0, c, false, false);
}

// A fragment (16x32 bf16) from row-major memory, K contiguous.
// lanes 0-15 (s=0): K 0..15 ; lanes 16-31 (s=1): K 16..31
__device__ __forceinline__ bf16x16 load_frag_a(const unsigned short* base, int s) {
  FragCvt f;
  f.q[0] = *reinterpret_cast<const uint4*>(base + s * 16);
  f.q[1] = *reinterpret_cast<const uint4*>(base + s * 16 + 8);
  return f.bf;
}

// B fragment (32x16 bf16) for NT gemm: per lane 16 contiguous K values
// (caller already added (col)*ldb + s*16 to base)
__device__ __forceinline__ bf16x16 load_frag_b(const unsigned short* base) {
  FragCvt f;
  f.q[0] = *reinterpret_cast<const uint4*>(base);
  f.q[1] = *reinterpret_cast<const uint4*>(base + 8);
  return f.bf;
}

// A fragment built from f32 logits in LDS: a = exp(logit - m), bf16; 0 beyond `limit`
__device__ __forceinline__ bf16x16 prob_frag(const float* row, int k0, int s, float m, int limit) {
  FragCvt f;
#pragma unroll
  for (int t = 0; t < 8; ++t) {
    int ka = k0 + s * 16 + t;
    int kb = ka + 8;
    f.us[t]     = (ka < limit) ? f2bf(__expf(row[ka] - m)) : (unsigned short)0;
    f.us[8 + t] = (kb < limit) ? f2bf(__expf(row[kb] - m)) : (unsigned short)0;
  }
  return f.bf;
}

// 32(M) x 64(N) output tile, NT gemm: C[m,n] = sum_k A[m,k]*B[n,k]
// Ping-pong double buffer, k-loop unrolled by 2 (K must be a multiple of 64):
// no register rotation moves, loads for step i+1 issued before WMMAs of step i.
__device__ __forceinline__ void gemm_tile_32x64(const unsigned short* __restrict__ A, int lda,
                                                const unsigned short* __restrict__ Bm, int ldb,
                                                int K, int m0, int n0, int lane, f32x8 acc[8]) {
  const int r = lane & 15, s = lane >> 4;
  const unsigned short* arow0 = A  + (size_t)(m0 + r) * lda;
  const unsigned short* arow1 = A  + (size_t)(m0 + 16 + r) * lda;
  const unsigned short* bbase = Bm + (size_t)(n0 + r) * ldb + s * 16;

  bf16x16 a0A = load_frag_a(arow0, s);
  bf16x16 a1A = load_frag_a(arow1, s);
  bf16x16 bA[4];
#pragma unroll
  for (int j = 0; j < 4; ++j) bA[j] = load_frag_b(bbase + (size_t)j * 16 * ldb);

  for (int k0 = 0; k0 < K; k0 += 64) {
    // prefetch k0+32 into B buffers (always in range: K % 64 == 0)
    bf16x16 a0B = load_frag_a(arow0 + k0 + 32, s);
    bf16x16 a1B = load_frag_a(arow1 + k0 + 32, s);
    bf16x16 bB[4];
#pragma unroll
    for (int j = 0; j < 4; ++j) bB[j] = load_frag_b(bbase + (size_t)j * 16 * ldb + k0 + 32);
#pragma unroll
    for (int j = 0; j < 4; ++j) acc[j]     = wmma_bf16(a0A, bA[j], acc[j]);
#pragma unroll
    for (int j = 0; j < 4; ++j) acc[4 + j] = wmma_bf16(a1A, bA[j], acc[4 + j]);
    if (k0 + 64 < K) {
      a0A = load_frag_a(arow0 + k0 + 64, s);
      a1A = load_frag_a(arow1 + k0 + 64, s);
#pragma unroll
      for (int j = 0; j < 4; ++j) bA[j] = load_frag_b(bbase + (size_t)j * 16 * ldb + k0 + 64);
    }
#pragma unroll
    for (int j = 0; j < 4; ++j) acc[j]     = wmma_bf16(a0B, bB[j], acc[j]);
#pragma unroll
    for (int j = 0; j < 4; ++j) acc[4 + j] = wmma_bf16(a1B, bB[j], acc[4 + j]);
  }
}

// ---------------- elementwise helpers ----------------
__global__ void cvt_bf16(const float* __restrict__ src, unsigned short* __restrict__ dst, int n) {
  int i = blockIdx.x * 256 + threadIdx.x;
  if (i < n) dst[i] = f2bf(src[i]);
}

// zero vT pad columns n in [819,832): garbage there would turn 0*x into NaN inside WMMA
__global__ void zero_vt_pads(unsigned short* __restrict__ vT) {
  int i = blockIdx.x * 256 + threadIdx.x;
  const int total = BATCH * HEADS * HD * (NPAD - NREAL);
  if (i < total) {
    int nn = i % (NPAD - NREAL);
    int bd = i / (NPAD - NREAL);
    vT[(size_t)bd * NPAD + NREAL + nn] = 0;
  }
}

// bias1 = bilinear(pb 4x4 -> 32x32, half-pixel, edge clamp); bias2 = pc + pr; xb = bf16(x+bias)
__global__ void bias_x_kernel(const float* __restrict__ x, const float* __restrict__ pb,
                              const float* __restrict__ pc, const float* __restrict__ pr,
                              unsigned short* __restrict__ xb) {
  int n = blockIdx.x;          // 0..818
  int t = threadIdx.x;         // 256
  const float* PB = pb + (size_t)n * 16;
  for (int c = t; c < DIMC; c += 256) {
    int i = c >> 5, j = c & 31;
    float fi = (i + 0.5f) * 0.125f - 0.5f;
    float fj = (j + 0.5f) * 0.125f - 0.5f;
    fi = fminf(fmaxf(fi, 0.f), 3.f);
    fj = fminf(fmaxf(fj, 0.f), 3.f);
    int i0 = (int)fi; if (i0 > 2) i0 = 2;
    int j0 = (int)fj; if (j0 > 2) j0 = 2;
    float wi = fi - (float)i0, wj = fj - (float)j0;
    float v00 = PB[i0 * 4 + j0],     v01 = PB[i0 * 4 + j0 + 1];
    float v10 = PB[(i0+1)*4 + j0],   v11 = PB[(i0+1)*4 + j0 + 1];
    float bias1 = (1.f - wi) * ((1.f - wj) * v00 + wj * v01)
                +        wi  * ((1.f - wj) * v10 + wj * v11);
    float bias = bias1 + pc[(size_t)n * 32 + i] + pr[(size_t)n * 32 + j];
#pragma unroll
    for (int b = 0; b < BATCH; ++b) {
      size_t idx = ((size_t)b * NREAL + n) * DIMC + c;
      xb[idx] = f2bf(x[idx] + bias);
    }
  }
}

// ---------------- GEMM kernels (bf16 WMMA, f32 accumulate) ----------------
__global__ void __launch_bounds__(128) gemm_qkv(const unsigned short* __restrict__ xb,
                                                const unsigned short* __restrict__ wq,
                                                unsigned short* __restrict__ qb,
                                                unsigned short* __restrict__ kb,
                                                unsigned short* __restrict__ vT) {
  int wt = blockIdx.x * 4 + (threadIdx.x >> 5);
  int lane = threadIdx.x & 31;
  int tn = wt % 48, tm = wt / 48;           // 205 x 48 tiles (32 x 64)
  int m0 = tm * 32, o0 = tn * 64;
  f32x8 acc[8] = {zero8(), zero8(), zero8(), zero8(), zero8(), zero8(), zero8(), zero8()};
  gemm_tile_32x64(xb, DIMC, wq, DIMC, DIMC, m0, o0, lane, acc);
  int r = lane & 15, s = lane >> 4;
#pragma unroll
  for (int mi = 0; mi < 2; ++mi) {
#pragma unroll
    for (int j = 0; j < 4; ++j) {
      const float* ap = (const float*)&acc[mi * 4 + j];
#pragma unroll
      for (int i = 0; i < 8; ++i) {
        int m = m0 + mi * 16 + i + 8 * s;
        if (m >= MREAL) continue;
        int o = o0 + j * 16 + r;
        int bb = m / NREAL, nn = m - bb * NREAL;
        int which = o >> 10, h = (o >> 6) & 15, d = o & 63;
        unsigned short v = f2bf(ap[i]);
        size_t bh = (size_t)bb * HEADS + h;
        if (which == 0)      qb[(bh * NPAD + nn) * HD + d] = v;   // q: (b,h,n,d)
        else if (which == 1) kb[(bh * NPAD + nn) * HD + d] = v;   // k: (b,h,n,d)
        else                 vT[(bh * HD + d) * NPAD + nn] = v;   // vT: (b,h,d,n)
      }
    }
  }
}

__global__ void __launch_bounds__(128) gemm_p(const unsigned short* __restrict__ px,
                                              const unsigned short* __restrict__ wp,
                                              const float* __restrict__ pbias,
                                              unsigned short* __restrict__ pbuf) {
  int wt = blockIdx.x * 4 + (threadIdx.x >> 5);
  int lane = threadIdx.x & 31;
  int tn = wt % 16, tm = wt / 16;           // 256 x 16 tiles (32 x 64)
  int m0 = tm * 32, o0 = tn * 64;
  f32x8 acc[8] = {zero8(), zero8(), zero8(), zero8(), zero8(), zero8(), zero8(), zero8()};
  gemm_tile_32x64(px, DIMC, wp, DIMC, DIMC, m0, o0, lane, acc);
  int r = lane & 15, s = lane >> 4;
#pragma unroll
  for (int mi = 0; mi < 2; ++mi) {
#pragma unroll
    for (int j = 0; j < 4; ++j) {
      const float* ap = (const float*)&acc[mi * 4 + j];
#pragma unroll
      for (int i = 0; i < 8; ++i) {
        int m = m0 + mi * 16 + i + 8 * s;
        int o = o0 + j * 16 + r;
        int bb = m >> 10, l = m & 1023;
        int h = o >> 6, d = o & 63;
        size_t bh = (size_t)bb * HEADS + h;
        pbuf[(bh * LLEN + l) * HD + d] = f2bf(ap[i] + pbias[o]);   // p: (b,h,l,d)
      }
    }
  }
}

__global__ void __launch_bounds__(128) gemm_proj(const unsigned short* __restrict__ o2,
                                                 const unsigned short* __restrict__ wo,
                                                 const float* __restrict__ obias,
                                                 float* __restrict__ out) {
  int wt = blockIdx.x * 4 + (threadIdx.x >> 5);
  int lane = threadIdx.x & 31;
  int tn = wt % 16, tm = wt / 16;           // 205 x 16 tiles (32 x 64)
  int m0 = tm * 32, o0 = tn * 64;
  f32x8 acc[8] = {zero8(), zero8(), zero8(), zero8(), zero8(), zero8(), zero8(), zero8()};
  gemm_tile_32x64(o2, DIMC, wo, DIMC, DIMC, m0, o0, lane, acc);
  int r = lane & 15, s = lane >> 4;
#pragma unroll
  for (int mi = 0; mi < 2; ++mi) {
#pragma unroll
    for (int j = 0; j < 4; ++j) {
      const float* ap = (const float*)&acc[mi * 4 + j];
#pragma unroll
      for (int i = 0; i < 8; ++i) {
        int m = m0 + mi * 16 + i + 8 * s;
        if (m >= MREAL) continue;
        int o = o0 + j * 16 + r;
        out[(size_t)m * DIMC + o] = ap[i] + obias[o];
      }
    }
  }
}

// ---------------- attention stage 1: proxy_v = softmax(p*S . k^T) . v ----------------
__global__ void __launch_bounds__(32) stage1_attn(const unsigned short* __restrict__ pbuf,
                                                  const unsigned short* __restrict__ kbuf,
                                                  const unsigned short* __restrict__ vTbuf,
                                                  unsigned short* __restrict__ pvT) {
  extern __shared__ float lds[];                 // 16 x NPAD logits + 2*16 reductions
  float* redm = lds + 16 * NPAD;
  float* reds = redm + 16;
  const int lane = threadIdx.x;
  const int r = lane & 15, s = lane >> 4;
  const int bh = blockIdx.y;
  const int l0 = blockIdx.x * 16;
  const unsigned short* P  = pbuf  + (size_t)bh * LLEN * HD;
  const unsigned short* K  = kbuf  + (size_t)bh * NPAD * HD;
  const unsigned short* VT = vTbuf + (size_t)bh * HD * NPAD;
  unsigned short* OUT = pvT + (size_t)bh * HD * LLEN;

  // phase A: logits tile (16 x NPAD) into LDS; ping-pong over n-tile pairs
  const unsigned short* prow = P + (size_t)(l0 + r) * HD;
  bf16x16 a0 = load_frag_a(prow, s);
  bf16x16 a1 = load_frag_a(prow + 32, s);
  const unsigned short* kp = K + (size_t)r * HD + s * 16;
  bf16x16 c0A = load_frag_b(kp);
  bf16x16 c1A = load_frag_b(kp + 32);
  for (int nt = 0; nt < NPAD / 16; nt += 2) {
    const unsigned short* kB = kp + (size_t)(nt + 1) * 16 * HD;
    bf16x16 c0B = load_frag_b(kB);
    bf16x16 c1B = load_frag_b(kB + 32);
    {
      f32x8 acc = zero8();
      acc = wmma_bf16(a0, c0A, acc);
      acc = wmma_bf16(a1, c1A, acc);
      const float* ap = (const float*)&acc;
#pragma unroll
      for (int i = 0; i < 8; ++i)
        lds[(size_t)(i + 8 * s) * NPAD + nt * 16 + r] = ap[i] * ATT_SCALE;
    }
    if (nt + 2 < NPAD / 16) {
      const unsigned short* kA = kp + (size_t)(nt + 2) * 16 * HD;
      c0A = load_frag_b(kA);
      c1A = load_frag_b(kA + 32);
    }
    {
      f32x8 acc = zero8();
      acc = wmma_bf16(a0, c0B, acc);
      acc = wmma_bf16(a1, c1B, acc);
      const float* ap = (const float*)&acc;
#pragma unroll
      for (int i = 0; i < 8; ++i)
        lds[(size_t)(i + 8 * s) * NPAD + (nt + 1) * 16 + r] = ap[i] * ATT_SCALE;
    }
  }
  __syncthreads();

  // phase B: per-row max & sum (n < NREAL only)
  for (int row = 0; row < 16; ++row) {
    float m = -1e30f;
    for (int j = lane; j < NREAL; j += 32) m = fmaxf(m, lds[(size_t)row * NPAD + j]);
#pragma unroll
    for (int off = 16; off > 0; off >>= 1) m = fmaxf(m, __shfl_xor(m, off, 32));
    float sum = 0.f;
    for (int j = lane; j < NREAL; j += 32) sum += __expf(lds[(size_t)row * NPAD + j] - m);
#pragma unroll
    for (int off = 16; off > 0; off >>= 1) sum += __shfl_xor(sum, off, 32);
    if (lane == 0) { redm[row] = m; reds[row] = 1.f / sum; }
  }
  __syncthreads();

  // phase C: (exp probs) . V; ping-pong B buffers, exp/LDS work overlaps loads
  float mrow = redm[r];
  f32x8 acc[4] = {zero8(), zero8(), zero8(), zero8()};
  const unsigned short* vb0 = VT + (size_t)r * NPAD + s * 16;
  bf16x16 bA[4];
#pragma unroll
  for (int dt = 0; dt < 4; ++dt) bA[dt] = load_frag_b(vb0 + (size_t)dt * 16 * NPAD);
  for (int k0 = 0; k0 < NPAD; k0 += 64) {
    bf16x16 bB[4];
#pragma unroll
    for (int dt = 0; dt < 4; ++dt) bB[dt] = load_frag_b(vb0 + (size_t)dt * 16 * NPAD + k0 + 32);
    bf16x16 aA = prob_frag(lds + (size_t)r * NPAD, k0, s, mrow, NREAL);
#pragma unroll
    for (int dt = 0; dt < 4; ++dt) acc[dt] = wmma_bf16(aA, bA[dt], acc[dt]);
    if (k0 + 64 < NPAD) {
#pragma unroll
      for (int dt = 0; dt < 4; ++dt) bA[dt] = load_frag_b(vb0 + (size_t)dt * 16 * NPAD + k0 + 64);
    }
    bf16x16 aB = prob_frag(lds + (size_t)r * NPAD, k0 + 32, s, mrow, NREAL);
#pragma unroll
    for (int dt = 0; dt < 4; ++dt) acc[dt] = wmma_bf16(aB, bB[dt], acc[dt]);
  }
#pragma unroll
  for (int dt = 0; dt < 4; ++dt) {
    const float* ap = (const float*)&acc[dt];
#pragma unroll
    for (int i = 0; i < 8; ++i) {
      int row = i + 8 * s;
      OUT[(size_t)(dt * 16 + r) * LLEN + l0 + row] = f2bf(ap[i] * reds[row]);
    }
  }
}

// ---------------- attention stage 2: out = softmax(mask(q*S . p^T)) . proxy_v ----------------
__global__ void __launch_bounds__(32) stage2_attn(const unsigned short* __restrict__ qbuf,
                                                  const unsigned short* __restrict__ pbuf,
                                                  const unsigned short* __restrict__ pvT,
                                                  const unsigned char* __restrict__ mask,
                                                  unsigned short* __restrict__ out2) {
  extern __shared__ float lds[];                 // 16 x LLEN logits + 2*16 reductions
  float* redm = lds + 16 * LLEN;
  float* reds = redm + 16;
  const int lane = threadIdx.x;
  const int r = lane & 15, s = lane >> 4;
  const int bh = blockIdx.y;
  const int b = bh >> 4, h = bh & 15;
  const int n0 = blockIdx.x * 16;
  const unsigned short* Q  = qbuf + (size_t)bh * NPAD * HD;
  const unsigned short* P  = pbuf + (size_t)bh * LLEN * HD;
  const unsigned short* PV = pvT  + (size_t)bh * HD * LLEN;
  const unsigned char*  mk = mask + (size_t)b * LLEN;

  // phase A: masked logits (16 x LLEN) into LDS; ping-pong over l-tile pairs
  const unsigned short* qrow = Q + (size_t)(n0 + r) * HD;
  bf16x16 a0 = load_frag_a(qrow, s);
  bf16x16 a1 = load_frag_a(qrow + 32, s);
  const unsigned short* pp = P + (size_t)r * HD + s * 16;
  bf16x16 c0A = load_frag_b(pp);
  bf16x16 c1A = load_frag_b(pp + 32);
  for (int lt = 0; lt < LLEN / 16; lt += 2) {
    const unsigned short* pB = pp + (size_t)(lt + 1) * 16 * HD;
    bf16x16 c0B = load_frag_b(pB);
    bf16x16 c1B = load_frag_b(pB + 32);
    {
      f32x8 acc = zero8();
      acc = wmma_bf16(a0, c0A, acc);
      acc = wmma_bf16(a1, c1A, acc);
      const float* ap = (const float*)&acc;
      int l = lt * 16 + r;
      bool mv = mk[l] != 0;
#pragma unroll
      for (int i = 0; i < 8; ++i)
        lds[(size_t)(i + 8 * s) * LLEN + l] = mv ? ap[i] * ATT_SCALE : -1e9f;
    }
    if (lt + 2 < LLEN / 16) {
      const unsigned short* pA = pp + (size_t)(lt + 2) * 16 * HD;
      c0A = load_frag_b(pA);
      c1A = load_frag_b(pA + 32);
    }
    {
      f32x8 acc = zero8();
      acc = wmma_bf16(a0, c0B, acc);
      acc = wmma_bf16(a1, c1B, acc);
      const float* ap = (const float*)&acc;
      int l = (lt + 1) * 16 + r;
      bool mv = mk[l] != 0;
#pragma unroll
      for (int i = 0; i < 8; ++i)
        lds[(size_t)(i + 8 * s) * LLEN + l] = mv ? ap[i] * ATT_SCALE : -1e9f;
    }
  }
  __syncthreads();

  for (int row = 0; row < 16; ++row) {
    float m = -1e30f;
    for (int j = lane; j < LLEN; j += 32) m = fmaxf(m, lds[(size_t)row * LLEN + j]);
#pragma unroll
    for (int off = 16; off > 0; off >>= 1) m = fmaxf(m, __shfl_xor(m, off, 32));
    float sum = 0.f;
    for (int j = lane; j < LLEN; j += 32) sum += __expf(lds[(size_t)row * LLEN + j] - m);
#pragma unroll
    for (int off = 16; off > 0; off >>= 1) sum += __shfl_xor(sum, off, 32);
    if (lane == 0) { redm[row] = m; reds[row] = 1.f / sum; }
  }
  __syncthreads();

  float mrow = redm[r];
  f32x8 acc[4] = {zero8(), zero8(), zero8(), zero8()};
  const unsigned short* pv0 = PV + (size_t)r * LLEN + s * 16;
  bf16x16 bA[4];
#pragma unroll
  for (int dt = 0; dt < 4; ++dt) bA[dt] = load_frag_b(pv0 + (size_t)dt * 16 * LLEN);
  for (int k0 = 0; k0 < LLEN; k0 += 64) {
    bf16x16 bB[4];
#pragma unroll
    for (int dt = 0; dt < 4; ++dt) bB[dt] = load_frag_b(pv0 + (size_t)dt * 16 * LLEN + k0 + 32);
    bf16x16 aA = prob_frag(lds + (size_t)r * LLEN, k0, s, mrow, LLEN);
#pragma unroll
    for (int dt = 0; dt < 4; ++dt) acc[dt] = wmma_bf16(aA, bA[dt], acc[dt]);
    if (k0 + 64 < LLEN) {
#pragma unroll
      for (int dt = 0; dt < 4; ++dt) bA[dt] = load_frag_b(pv0 + (size_t)dt * 16 * LLEN + k0 + 64);
    }
    bf16x16 aB = prob_frag(lds + (size_t)r * LLEN, k0 + 32, s, mrow, LLEN);
#pragma unroll
    for (int dt = 0; dt < 4; ++dt) acc[dt] = wmma_bf16(aB, bB[dt], acc[dt]);
  }
#pragma unroll
  for (int dt = 0; dt < 4; ++dt) {
    const float* ap = (const float*)&acc[dt];
#pragma unroll
    for (int i = 0; i < 8; ++i) {
      int n = n0 + i + 8 * s;
      if (n < NREAL) {
        int d = dt * 16 + r;
        out2[((size_t)b * NREAL + n) * DIMC + (size_t)h * HD + d] = f2bf(ap[i] * reds[i + 8 * s]);
      }
    }
  }
}

// ---------------- host launcher ----------------
extern "C" void kernel_launch(void* const* d_in, const int* in_sizes, int n_in,
                              void* d_out, int out_size, void* d_ws, size_t ws_size,
                              hipStream_t stream) {
  (void)in_sizes; (void)n_in; (void)out_size; (void)ws_size;
  const float* x        = (const float*)d_in[0];
  const float* proxy    = (const float*)d_in[1];
  const unsigned char* mask = (const unsigned char*)d_in[2];
  const float* qkv_w    = (const float*)d_in[3];
  const float* proxy_w  = (const float*)d_in[4];
  const float* proxy_b  = (const float*)d_in[5];
  const float* proj_w   = (const float*)d_in[6];
  const float* proj_b   = (const float*)d_in[7];
  const float* pb_bias  = (const float*)d_in[8];
  const float* pc_bias  = (const float*)d_in[9];
  const float* pr_bias  = (const float*)d_in[10];
  float* out = (float*)d_out;

  char* ws = (char*)d_ws;
  auto alloc = [&](size_t bytes) {
    char* p = ws;
    ws += (bytes + 255) & ~(size_t)255;
    return (unsigned short*)p;
  };
  unsigned short* xb   = alloc((size_t)MQ * DIMC * 2);                    // bf16 x + bias
  unsigned short* wq   = alloc((size_t)3 * DIMC * DIMC * 2);              // bf16 qkv_w
  unsigned short* wp   = alloc((size_t)DIMC * DIMC * 2);                  // bf16 proxy_w
  unsigned short* wo   = alloc((size_t)DIMC * DIMC * 2);                  // bf16 proj_w
  unsigned short* px   = alloc((size_t)BATCH * LLEN * DIMC * 2);          // bf16 proxy
  unsigned short* qb   = alloc((size_t)BATCH * HEADS * NPAD * HD * 2);    // q  (b,h,n,d)
  unsigned short* kb   = alloc((size_t)BATCH * HEADS * NPAD * HD * 2);    // k  (b,h,n,d)
  unsigned short* vT   = alloc((size_t)BATCH * HEADS * HD * NPAD * 2);    // vT (b,h,d,n)
  unsigned short* pbuf = alloc((size_t)BATCH * HEADS * LLEN * HD * 2);    // p  (b,h,l,d)
  unsigned short* pvT  = alloc((size_t)BATCH * HEADS * HD * LLEN * 2);    // pvT(b,h,d,l)
  unsigned short* o2   = alloc((size_t)MQ * DIMC * 2);                    // stage-2 out (b,n,c)

  // 1) weight / proxy conversions to bf16
  {
    int n;
    n = 3 * DIMC * DIMC;      cvt_bf16<<<(n + 255) / 256, 256, 0, stream>>>(qkv_w, wq, n);
    n = DIMC * DIMC;          cvt_bf16<<<(n + 255) / 256, 256, 0, stream>>>(proxy_w, wp, n);
    n = DIMC * DIMC;          cvt_bf16<<<(n + 255) / 256, 256, 0, stream>>>(proj_w, wo, n);
    n = BATCH * LLEN * DIMC;  cvt_bf16<<<(n + 255) / 256, 256, 0, stream>>>(proxy, px, n);
  }
  // 2) zero vT pads (only pad feeding a contraction)
  {
    int n = BATCH * HEADS * HD * (NPAD - NREAL);
    zero_vt_pads<<<(n + 255) / 256, 256, 0, stream>>>(vT);
  }
  // 3) bias + x -> bf16
  bias_x_kernel<<<NREAL, 256, 0, stream>>>(x, pb_bias, pc_bias, pr_bias, xb);
  // 4) qkv GEMM, scatter to q/k/vT   (205*48 tiles of 32x64, 4 waves per block)
  gemm_qkv<<<(205 * 48) / 4, 128, 0, stream>>>(xb, wq, qb, kb, vT);
  // 5) proxy projection GEMM -> p    (256*16 tiles)
  gemm_p<<<(256 * 16) / 4, 128, 0, stream>>>(px, wp, proxy_b, pbuf);
  // 6) stage-1 attention: one wave per 16-row L tile
  {
    dim3 grid(LLEN / 16, BATCH * HEADS);
    size_t shmem = (size_t)16 * NPAD * 4 + 32 * 4;
    stage1_attn<<<grid, 32, shmem, stream>>>(pbuf, kb, vT, pvT);
  }
  // 7) stage-2 attention: one wave per 16-row N tile
  {
    dim3 grid(NPAD / 16, BATCH * HEADS);
    size_t shmem = (size_t)16 * LLEN * 4 + 32 * 4;
    stage2_attn<<<grid, 32, shmem, stream>>>(qb, pbuf, pvT, mask, o2);
  }
  // 8) output projection GEMM -> fp32 out  (205*16 tiles)
  gemm_proj<<<(205 * 16) / 4, 128, 0, stream>>>(o2, wo, proj_b, out);
}
